// EnEquivariantMLP_50225347559547
// MI455X (gfx1250) — matrix operand order repaired and checked by reading, not compiled, not run on hardware
//
#include <hip/hip_runtime.h>

typedef __attribute__((ext_vector_type(16))) _Float16 v16h;
typedef __attribute__((ext_vector_type(8)))  _Float16 v8h;
typedef __attribute__((ext_vector_type(8)))  float    v8f;
typedef __attribute__((ext_vector_type(4)))  float    v4f;
typedef __attribute__((ext_vector_type(4)))  unsigned int v4u;
typedef __attribute__((ext_vector_type(8)))  int      v8i;
typedef __attribute__((ext_vector_type(4)))  int      v4i;

#define NN    200000
#define DDIM  3
#define FIN   128
#define FMID  512
#define FOUT  128

// d_ws layout in _Float16 units. W1/W2 stored TRANSPOSED [col][k] with the
// K-dimension split into 16-half blocks swizzled by (col&7) so the LDS image
// (copied linearly by the TDM) is ~bank-conflict-free for B-fragment loads.
#define WS_WP 0                      // 128x128, linear [n][k]
#define WS_W1 (FIN * FOUT)           // [n<512][k<128], block-swizzled
#define WS_W2 (WS_W1 + FIN * FMID)   // [n<128][k<512], block-swizzled

// dynamic LDS layout (halves)
#define LDS_W1 0
#define LDS_W2 (FIN * FMID)                 // 65536
#define LDS_H1 (LDS_W2 + FMID * FOUT)       // 131072  (+wave*1024)
#define LDS_H2 (LDS_H1 + 8 * 16 * 64)       // 139264  (+wave*2048)
#define SMEM_HALVES (LDS_H2 + 8 * 16 * FOUT)
#define SMEM_BYTES  (SMEM_HALVES * 2)       // 311296

#if defined(__has_builtin)
#if __has_builtin(__builtin_amdgcn_tensor_load_to_lds) && \
    __has_builtin(__builtin_amdgcn_s_wait_tensorcnt)
#define USE_TDM 1
#endif
#endif

// ---------------------------------------------------------------------------
// Prologue: fp32 weights -> f16 in d_ws, transposed + block-swizzled.
// ---------------------------------------------------------------------------
__global__ __launch_bounds__(256) void convert_weights_f16t(
    const float* __restrict__ Wp, const float* __restrict__ W1,
    const float* __restrict__ W2, _Float16* __restrict__ ws) {
  int i = blockIdx.x * blockDim.x + threadIdx.x;
  const int nWp = FIN * FOUT, nW1 = FIN * FMID, nW2 = FMID * FOUT;
  if (i < nWp) {
    int n = i >> 7, k = i & 127;                       // Wp_t[n][k] = Wp[k][n]
    ws[WS_WP + i] = (_Float16)Wp[k * FOUT + n];
  } else if (i < nWp + nW1) {
    int t = i - nWp;
    int n = t >> 7, r = t & 127;                       // swizzled W1t
    int bs = r >> 4, w = r & 15;
    int k = ((bs ^ (n & 7)) << 4) + w;
    ws[WS_W1 + t] = (_Float16)W1[k * FMID + n];
  } else if (i < nWp + nW1 + nW2) {
    int t = i - nWp - nW1;
    int n = t >> 9, r = t & 511;                       // swizzled W2t
    int bs = r >> 4, w = r & 15;
    int k = ((bs ^ (n & 7)) << 4) + w;
    ws[WS_W2 + t] = (_Float16)W2[k * FOUT + n];
  }
}

__device__ __forceinline__ v16h combine8(v8h lo, v8h hi2) {
  v16h f;
  #pragma unroll
  for (int e = 0; e < 8; ++e) { f[e] = lo[e]; f[e + 8] = hi2[e]; }
  return f;
}

// B-fragment from linear global memory (W_prop path)
__device__ __forceinline__ v16h load_bfrag_g(const _Float16* p) {
  const v8h* q = (const v8h*)p;
  return combine8(q[0], q[1]);
}

// B-fragment from swizzled LDS weight image: one 16-half K-block per fragment.
__device__ __forceinline__ v16h load_bfrag_lds(const _Float16* base, int col,
                                               int kdim, int blk) {
  const _Float16* p = base + col * kdim + ((blk ^ (col & 7)) << 4);
  const v8h* q = (const v8h*)p;
  return combine8(q[0], q[1]);
}

#ifdef USE_TDM
// 1-D TDM copy: bytes (multiple of 8) from global gaddr -> LDS offset lds_off.
__device__ __forceinline__ void tdm_load_1d(unsigned lds_off, const void* gaddr,
                                            unsigned bytes) {
  unsigned long long ga = (unsigned long long)gaddr;
  unsigned n8 = bytes >> 3;                       // 8-byte elements
  v4u g0;
  g0[0] = 1u;                                     // count=1, user-mode load
  g0[1] = lds_off;                                // LDS byte address
  g0[2] = (unsigned)(ga & 0xffffffffu);           // global_addr[31:0]
  g0[3] = (unsigned)((ga >> 32) & 0x1ffffffu) | (2u << 30);  // addr hi | type=2
  v8i g1;
  g1[0] = (int)(3u << 16);                        // data_size=3 (8B), no mask
  g1[1] = (int)((n8 & 0xffffu) << 16);            // tensor_dim0[15:0]
  g1[2] = (int)((n8 >> 16) & 0xffffu) | (1 << 16);// tensor_dim0[31:16], dim1=1
  g1[3] = (int)((n8 & 0xffffu) << 16);            // tile_dim0 = n8
  g1[4] = 0;                                      // tile_dim1/2 unused
  g1[5] = (int)n8;                                // tensor_dim0_stride lo
  g1[6] = 0;                                      // stride0 hi, stride1 lo
  g1[7] = 0;
  v4i g2; g2[0] = 1; g2[1] = 1; g2[2] = (int)n8; g2[3] = 0;  // dims2/3 = 1
  v4i g3; g3[0] = (int)n8; g3[1] = (1 << 16); g3[2] = 0; g3[3] = 0;
  v8i g4; // unused trailing descriptor group (VADDR4 slot) -- zeros
  #pragma unroll
  for (int z = 0; z < 8; ++z) g4[z] = 0;
  __builtin_amdgcn_tensor_load_to_lds(g0, g1, g2, g3, g4, 0);
}
#endif

// ---------------------------------------------------------------------------
// Fused kernel: one wave32 per 16 nodes, 8 waves per block.
//   cfrag = sum_d x^2  (A-fragments straight from global x)
//   h2    = tanh(c@W1+b1)@W2+b2  (W1/W2 from block-shared LDS via TDM stage)
//   out   = (x @ W_prop) * h2
// ---------------------------------------------------------------------------
__global__ __launch_bounds__(256) void fused_equivariant_mlp(
    const float* __restrict__ x, const float* __restrict__ b1,
    const float* __restrict__ b2, const _Float16* __restrict__ ws,
    float* __restrict__ out) {
  extern __shared__ _Float16 smem[];

  const int tid  = threadIdx.x;
  const int wave = tid >> 5;
  const int lane = tid & 31;
  const int lm   = lane & 15;     // row (A) / col (B,C) index within tile
  const int hi   = lane >> 4;     // lane-half selector
  const int nodeBase = blockIdx.x * 128 + wave * 16;

  _Float16* ldsW1 = smem + LDS_W1;
  _Float16* ldsW2 = smem + LDS_W2;
  _Float16* h1buf = smem + LDS_H1 + wave * (16 * 64);
  _Float16* h2buf = smem + LDS_H2 + wave * (16 * FOUT);

  // ---------- stage W1t+W2t (contiguous 256KB) into LDS ----------
#ifdef USE_TDM
  if (wave == 0) {
    tdm_load_1d((unsigned)(size_t)(void*)ldsW1, ws + WS_W1,
                (FIN * FMID + FMID * FOUT) * 2u);
    __builtin_amdgcn_s_wait_tensorcnt(0);
  }
#else
  {
    const v8h* src = (const v8h*)(ws + WS_W1);
    v8h* dst = (v8h*)ldsW1;
    for (int i = tid; i < (FIN * FMID + FMID * FOUT) / 8; i += 256)
      dst[i] = src[i];
  }
#endif
  __syncthreads();

  // ---------- c A-fragments: c[m][k] = sum_d x[m][d][k]^2 (f16) ----------
  v16h cfrag[4];
  {
    int node = nodeBase + lm; if (node >= NN) node = NN - 1;
    const float* xr = x + (size_t)node * (DDIM * FIN);
    __builtin_prefetch(xr, 0, 1);
    __builtin_prefetch(xr + 2 * FIN, 0, 1);
    #pragma unroll
    for (int kb = 0; kb < 4; ++kb) {
      #pragma unroll
      for (int run = 0; run < 2; ++run) {
        const int K0 = kb * 32 + hi * 16 + run * 8;   // A-layout run of 8 K
        v4f s0, s1;
        #pragma unroll
        for (int i = 0; i < 4; ++i) { s0[i] = 0.f; s1[i] = 0.f; }
        #pragma unroll
        for (int d = 0; d < DDIM; ++d) {
          v4f a = *(const v4f*)(xr + d * FIN + K0);
          v4f b = *(const v4f*)(xr + d * FIN + K0 + 4);
          #pragma unroll
          for (int i = 0; i < 4; ++i) { s0[i] += a[i] * a[i]; s1[i] += b[i] * b[i]; }
        }
        #pragma unroll
        for (int i = 0; i < 4; ++i) {
          cfrag[kb][run * 8 + i]     = (_Float16)s0[i];
          cfrag[kb][run * 8 + 4 + i] = (_Float16)s1[i];
        }
      }
    }
  }

  // ---------- h2 accumulators (8 col tiles of 16), init with b2 ----------
  v8f acc2[8];
  #pragma unroll
  for (int n = 0; n < 8; ++n) {
    float bv = b2[n * 16 + lm];
    #pragma unroll
    for (int r = 0; r < 8; ++r) acc2[n][r] = bv;
  }

  // ---------- fused GEMM2 -> tanh -> GEMM3 over F_MID chunks of 64 ----------
  const int hsw = (lm >> 1) & 3;                 // h1 bounce swizzle
  #pragma unroll 1
  for (int j = 0; j < 8; ++j) {
    #pragma unroll
    for (int u = 0; u < 4; ++u) {
      const int col = j * 64 + u * 16 + lm;
      v8f acc1;
      const float bv = b1[col];
      #pragma unroll
      for (int r = 0; r < 8; ++r) acc1[r] = bv;
      #pragma unroll
      for (int kb = 0; kb < 4; ++kb) {
        v16h bf = load_bfrag_lds(ldsW1, col, FIN, kb * 2 + hi);
        acc1 = __builtin_amdgcn_wmma_f32_16x16x32_f16(false, cfrag[kb], false, bf,
                                                      (short)0, acc1, false, false);
      }
      // tanh + C->A relayout through LDS (swizzled 16-half blocks)
      #pragma unroll
      for (int r = 0; r < 8; ++r) {
        const int m = r + hi * 8;
        h1buf[m * 64 + ((u ^ ((m >> 1) & 3)) << 4) + lm] = (_Float16)tanhf(acc1[r]);
      }
    }
    asm volatile("s_wait_dscnt 0" ::: "memory");  // cross-lane LDS handoff
    v16h hfrag[2];
    #pragma unroll
    for (int ks = 0; ks < 2; ++ks) {
      hfrag[ks] = combine8(
          *(const v8h*)(h1buf + lm * 64 + (((ks * 2 + hi) ^ hsw) << 4)),
          *(const v8h*)(h1buf + lm * 64 + (((ks * 2 + hi) ^ hsw) << 4) + 8));
    }
    #pragma unroll
    for (int n = 0; n < 8; ++n) {
      const int ncol = n * 16 + lm;
      #pragma unroll
      for (int ks = 0; ks < 2; ++ks) {
        v16h bf = load_bfrag_lds(ldsW2, ncol, FMID, j * 4 + ks * 2 + hi);
        acc2[n] = __builtin_amdgcn_wmma_f32_16x16x32_f16(false, hfrag[ks], false, bf,
                                                         (short)0, acc2[n], false, false);
      }
    }
    asm volatile("s_wait_dscnt 0" ::: "memory");
  }

  // ---------- stash gate h2 in LDS (f16, row-major [m][g]) ----------
  #pragma unroll
  for (int n = 0; n < 8; ++n) {
    #pragma unroll
    for (int r = 0; r < 8; ++r)
      h2buf[(r + hi * 8) * FOUT + n * 16 + lm] = (_Float16)acc2[n][r];
  }
  asm volatile("s_wait_dscnt 0" ::: "memory");

  // ---------- GEMM1: (x @ W_prop) * h2; W_prop frags reused across 3 tiles --
  v16h af[3][4];
  #pragma unroll
  for (int t = 0; t < 3; ++t) {
    const int row = t * 16 + lm;                 // flattened (node,dim) row
    const int nl = row / 3, dc = row - nl * 3;
    int node = nodeBase + nl; if (node >= NN) node = NN - 1;
    const float* xr = x + ((size_t)node * DDIM + dc) * FIN;
    #pragma unroll
    for (int kb = 0; kb < 4; ++kb) {
      #pragma unroll
      for (int run = 0; run < 2; ++run) {
        const int K0 = kb * 32 + hi * 16 + run * 8;
        v4f a = *(const v4f*)(xr + K0);
        v4f b = *(const v4f*)(xr + K0 + 4);
        #pragma unroll
        for (int i = 0; i < 4; ++i) {
          af[t][kb][run * 8 + i]     = (_Float16)a[i];
          af[t][kb][run * 8 + 4 + i] = (_Float16)b[i];
        }
      }
    }
  }
  #pragma unroll 1
  for (int n = 0; n < 8; ++n) {
    v16h wpf[4];
    const _Float16* wpc = ws + WS_WP + (size_t)(n * 16 + lm) * FIN;
    #pragma unroll
    for (int kb = 0; kb < 4; ++kb)
      wpf[kb] = load_bfrag_g(wpc + kb * 32 + hi * 16);
    #pragma unroll
    for (int t = 0; t < 3; ++t) {
      v8f acc;
      #pragma unroll
      for (int r = 0; r < 8; ++r) acc[r] = 0.f;
      #pragma unroll
      for (int kb = 0; kb < 4; ++kb)
        acc = __builtin_amdgcn_wmma_f32_16x16x32_f16(false, af[t][kb], false, wpf[kb],
                                                     (short)0, acc, false, false);
      #pragma unroll
      for (int r = 0; r < 8; ++r) {
        const int rg = t * 16 + r + hi * 8;          // 0..47
        const int m2 = rg / 3, d2 = rg - m2 * 3;
        const int node2 = nodeBase + m2;
        if (node2 < NN) {
          const float g = (float)h2buf[m2 * FOUT + n * 16 + lm];
          out[((size_t)node2 * DDIM + d2) * FOUT + n * 16 + lm] = acc[r] * g;
        }
      }
    }
  }
}

// ---------------------------------------------------------------------------
extern "C" void kernel_launch(void* const* d_in, const int* in_sizes, int n_in,
                              void* d_out, int out_size, void* d_ws, size_t ws_size,
                              hipStream_t stream) {
  (void)in_sizes; (void)n_in; (void)out_size; (void)ws_size;
  const float* x  = (const float*)d_in[0];
  const float* Wp = (const float*)d_in[1];
  const float* W1 = (const float*)d_in[2];
  const float* b1 = (const float*)d_in[3];
  const float* W2 = (const float*)d_in[4];
  const float* b2 = (const float*)d_in[5];
  _Float16* ws = (_Float16*)d_ws;
  float* out = (float*)d_out;

  static_assert(SMEM_BYTES == 311296, "lds budget");
  (void)hipFuncSetAttribute((const void*)fused_equivariant_mlp,
                            hipFuncAttributeMaxDynamicSharedMemorySize,
                            SMEM_BYTES);

  const int totalW = FIN * FOUT + FIN * FMID + FMID * FOUT;  // 147456
  convert_weights_f16t<<<(totalW + 255) / 256, 256, 0, stream>>>(Wp, W1, W2, ws);

  const int nblocks = (NN + 127) / 128;                      // 1563
  fused_equivariant_mlp<<<nblocks, 256, SMEM_BYTES, stream>>>(x, b1, b2, ws, out);
}
